// MultiHeadAttention_53051436040680
// MI455X (gfx1250) — matrix operand order repaired and checked by reading, not compiled, not run on hardware
//
#include <hip/hip_runtime.h>
#include <cmath>
#include <cstdint>

// ---------------------------------------------------------------------------
// CDNA5 (gfx1250) fused MHA forward: bf16 WMMA 16x16x32, f32 accumulate.
// Wave32 only. Fragments assembled as 8x u32 words (2x ds_load_b128).
// K/V attention tiles arrive via double-buffered GLOBAL_LOAD_ASYNC_TO_LDS_B128
// (ASYNCcnt), with V pre-transposed in the workspace so both tiles are pure
// contiguous async copies.
// ---------------------------------------------------------------------------

#if defined(__has_builtin)
#  if __has_builtin(__builtin_amdgcn_global_load_async_to_lds_b128)
#    define HAVE_ASYNC_LDS 1
#  endif
#endif

typedef __bf16   v16bf __attribute__((ext_vector_type(16)));
typedef float    v8f   __attribute__((ext_vector_type(8)));
typedef uint32_t v8u   __attribute__((ext_vector_type(8)));
typedef int      v4i   __attribute__((ext_vector_type(4)));

union frag_cast { v8u u; v16bf bf; };

static constexpr int Bn = 2, Sn = 2048, En = 1024, Hn = 16, Dn = 64;
static constexpr int Rn = Bn * Sn;  // 4096 token rows

__device__ __forceinline__ uint32_t pk_bf16(float lo, float hi) {
  union { __bf16 h[2]; uint32_t u; } p;
  p.h[0] = (__bf16)lo;
  p.h[1] = (__bf16)hi;
  return p.u;  // v_cvt_pk_bf16_f32
}

// A-matrix 16x32 bf16 fragment (ISA §7.12.2): lane holds row m=lane&15,
// half lh selects K {lh*8..+7} and {16+lh*8..+7}.
__device__ __forceinline__ v16bf load_a_frag_u(const __bf16* src, int lda, int lane) {
  const int m = lane & 15, lh = lane >> 4;
  const uint32_t* p = (const uint32_t*)(src + m * lda);
  frag_cast f;
#pragma unroll
  for (int j = 0; j < 4; ++j) f.u[j] = p[lh * 4 + j];
#pragma unroll
  for (int j = 0; j < 4; ++j) f.u[4 + j] = p[8 + lh * 4 + j];
  return f.bf;
}

// B-matrix 32x16 bf16 fragment from n-major storage: element (k,n) at
// bf16 src[n*2*ldu + k]; u[j] = pu[n*ldu + lh*4 + j], u[4+j] = pu[n*ldu + 8 + lh*4 + j].
__device__ __forceinline__ v16bf load_b_nmaj_u(const uint32_t* pu, int ldu, int lane) {
  const int n = lane & 15, lh = lane >> 4;
  frag_cast f;
#pragma unroll
  for (int j = 0; j < 4; ++j) f.u[j] = pu[n * ldu + lh * 4 + j];
#pragma unroll
  for (int j = 0; j < 4; ++j) f.u[4 + j] = pu[n * ldu + 8 + lh * 4 + j];
  return f.bf;
}

__device__ __forceinline__ v8f wmma_bf16(v16bf a, v16bf b, v8f c) {
  return __builtin_amdgcn_wmma_f32_16x16x32_bf16(false, a, false, b, (short)0, c,
                                                 false, false);
}

// 16-byte global->LDS copy: async (ASYNCcnt) when available, sync fallback.
__device__ __forceinline__ void copy16_g2l(const __bf16* g, __bf16* l) {
#if defined(HAVE_ASYNC_LDS)
  __builtin_amdgcn_global_load_async_to_lds_b128(
      (__attribute__((address_space(1))) v4i*)(g),
      (__attribute__((address_space(3))) v4i*)(l), 0, 0);
#else
  *(uint4*)l = *(const uint4*)g;
#endif
}

// Wait until <= N async copies outstanding (no-op on sync fallback).
__device__ __forceinline__ void wait_async_le0() {
#if defined(HAVE_ASYNC_LDS)
  asm volatile("s_wait_asynccnt 0" ::: "memory");
#endif
}
__device__ __forceinline__ void wait_async_le1() {
#if defined(HAVE_ASYNC_LDS)
  asm volatile("s_wait_asynccnt 1" ::: "memory");
#endif
}
__device__ __forceinline__ void wait_async_le2() {
#if defined(HAVE_ASYNC_LDS)
  asm volatile("s_wait_asynccnt 2" ::: "memory");
#endif
}

// ---------------------------------------------------------------------------
// Kernel 1: Y = (X @ W + bias) * scale -> bf16 workspace.
// TMODE 0: head-major [B,H,S,D] (Q,K). TMODE 1: transposed [B,H,D,S] (V).
// Compile-time TMODE -> straight-line epilogue (no per-store branches).
// Block = 256 threads (8 waves), 64x64 tile, K-step 32; 2 WMMAs/wave/step.
// ---------------------------------------------------------------------------
template <int TMODE>
__global__ __launch_bounds__(256) void qkv_proj_kernel(
    const float* __restrict__ X, const float* __restrict__ W,
    const float* __restrict__ bias, __bf16* __restrict__ Obf, float scale) {
  __shared__ __align__(16) __bf16 As[64 * 32];   // [m][k], lda 32
  __shared__ __align__(16) __bf16 BsT[64 * 32];  // [n][k], transposed, ld 32
  const int tid  = threadIdx.x;
  const int lane = tid & 31, w = tid >> 5;
  const int rbase = blockIdx.y * 64;
  const int cbase = blockIdx.x * 64;
  const int rsub  = (w & 3) << 4;
  const int nsub  = (w >> 2) << 5;

  v8f acc0 = {}, acc1 = {};
  for (int k0 = 0; k0 < En; k0 += 32) {
    // A tile: 64x32 f32 -> bf16; float4 loads, packed u32 stores.
#pragma unroll
    for (int it = 0; it < 2; ++it) {
      const int i = tid + it * 256;
      const int rr = i >> 3, q = i & 7;
      const float4 x = ((const float4*)(X + (size_t)(rbase + rr) * En + k0))[q];
      uint32_t* dst = (uint32_t*)As + rr * 16 + q * 2;
      dst[0] = pk_bf16(x.x, x.y);
      dst[1] = pk_bf16(x.z, x.w);
    }
    // B tile transposed: BsT[n][k] <- W[k0+k][cbase+n]; pack k-pairs.
#pragma unroll
    for (int it = 0; it < 4; ++it) {
      const int i = tid + it * 256;
      const int nn = i & 63, kp = i >> 6;
      const float w0 = W[(size_t)(k0 + 2 * kp) * En + cbase + nn];
      const float w1 = W[(size_t)(k0 + 2 * kp + 1) * En + cbase + nn];
      ((uint32_t*)BsT)[nn * 16 + kp] = pk_bf16(w0, w1);
    }
    __syncthreads();
    const v16bf a  = load_a_frag_u(As + rsub * 32, 32, lane);
    const v16bf b0 = load_b_nmaj_u((const uint32_t*)BsT + nsub * 16, 16, lane);
    const v16bf b1 = load_b_nmaj_u((const uint32_t*)BsT + (nsub + 16) * 16, 16, lane);
    acc0 = wmma_bf16(a, b0, acc0);
    acc1 = wmma_bf16(a, b1, acc1);
    __syncthreads();
  }

  const int n = lane & 15, lh = lane >> 4;
#pragma unroll
  for (int v = 0; v < 8; ++v) {
    const int r = rbase + rsub + v + 8 * lh;
    const int bb = r >> 11, ss = r & (Sn - 1);
#pragma unroll
    for (int t = 0; t < 2; ++t) {
      const int c = cbase + nsub + t * 16 + n;
      const float val = ((t ? acc1 : acc0)[v] + bias[c]) * scale;
      const int hh = c >> 6, dd = c & (Dn - 1);
      if (TMODE == 0)
        Obf[(((size_t)bb * Hn + hh) * Sn + ss) * Dn + dd] = (__bf16)val;
      else
        Obf[(((size_t)bb * Hn + hh) * Dn + dd) * Sn + ss] = (__bf16)val;
    }
  }
}

// ---------------------------------------------------------------------------
// Kernel 2: causal flash attention per (b,h). Block = 256 threads (8 waves),
// 128 query rows. K [S,D] and V^T [D,S] tiles are double-buffered async
// copies into LDS; each wave issues 2 asyncs per tile, so waiting
// asynccnt<=2 drains the current tile while the next stays in flight.
// ---------------------------------------------------------------------------
__global__ __launch_bounds__(256) void attn_kernel(
    const __bf16* __restrict__ Qb, const __bf16* __restrict__ Kb,
    const __bf16* __restrict__ VbT, __bf16* __restrict__ Cb) {
  __shared__ __align__(16) __bf16 Ks[2][32 * 64];   // [key][d]  (n-major QK^T)
  __shared__ __align__(16) __bf16 Vt[2][64 * 32];   // [d][key]  (n-major PV)
  __shared__ __align__(16) __bf16 Ps[8][16 * 32];   // wave-private P, lda 32

  const int tid  = threadIdx.x;
  const int lane = tid & 31, w = tid >> 5;
  const int bh = blockIdx.y;               // b*H + h
  const int bb = bh / Hn, hh = bh % Hn;
  const int qblk = blockIdx.x * 128;
  const int q0   = qblk + w * 16;
  const int qhi  = q0 + 15;

  const __bf16* Qh  = Qb  + (size_t)bh * Sn * Dn;
  const __bf16* Kh  = Kb  + (size_t)bh * Sn * Dn;
  const __bf16* VhT = VbT + (size_t)bh * Sn * Dn;  // [D][S]

  const int n = lane & 15, lh = lane >> 4;
  const int rrV = tid >> 2, qV = tid & 3;          // V^T tile chunk coords

  const v16bf aQ0 = load_a_frag_u(Qh + (size_t)q0 * Dn, Dn, lane);       // d 0..31
  const v16bf aQ1 = load_a_frag_u(Qh + (size_t)q0 * Dn + 32, Dn, lane);  // d 32..63

  v8f acc0 = {}, acc1 = {}, acc2 = {}, acc3 = {};
  float mrow[8], lrow[8];
#pragma unroll
  for (int v = 0; v < 8; ++v) { mrow[v] = -INFINITY; lrow[v] = 0.0f; }

  const int nTiles = (qblk + 127) / 32 + 1;        // uniform across block

  // Issue tile 0 (2 async b128 copies per thread: one K chunk, one V chunk).
  copy16_g2l(Kh + (size_t)0 * Dn + tid * 8, Ks[0] + tid * 8);
  copy16_g2l(VhT + (size_t)rrV * Sn + 0 + qV * 8, Vt[0] + rrV * 32 + qV * 8);

  for (int tIdx = 0; tIdx < nTiles; ++tIdx) {
    const int j0  = tIdx * 32;
    const int buf = tIdx & 1;
    const bool hasNext = (tIdx + 1) < nTiles;
    if (hasNext) {  // prefetch next tile into the other buffer
      const int j1 = j0 + 32;
      copy16_g2l(Kh + (size_t)j1 * Dn + tid * 8, Ks[buf ^ 1] + tid * 8);
      copy16_g2l(VhT + (size_t)rrV * Sn + j1 + qV * 8, Vt[buf ^ 1] + rrV * 32 + qV * 8);
      wait_async_le2();   // current tile complete; next still in flight
    } else {
      wait_async_le0();
    }
    __syncthreads();

    if (j0 <= qhi) {  // wave-uniform: EXEC all-ones around WMMA
      const uint32_t* ku = (const uint32_t*)Ks[buf];
      v8f s0 = {}, s1 = {};
      s0 = wmma_bf16(aQ0, load_b_nmaj_u(ku + 0,            32, lane), s0);
      s0 = wmma_bf16(aQ1, load_b_nmaj_u(ku + 16,           32, lane), s0);
      s1 = wmma_bf16(aQ0, load_b_nmaj_u(ku + 16 * 32,      32, lane), s1);
      s1 = wmma_bf16(aQ1, load_b_nmaj_u(ku + 16 * 32 + 16, 32, lane), s1);

      // Causal mask (both softmax scalings folded into Q at projection).
#pragma unroll
      for (int v = 0; v < 8; ++v) {
        const int qi = q0 + v + 8 * lh;
        if (j0 + n > qi)      s0[v] = -INFINITY;
        if (j0 + 16 + n > qi) s1[v] = -INFINITY;
      }

      // Online softmax: row stats via half-wave (16-lane) shuffles.
#pragma unroll
      for (int v = 0; v < 8; ++v) {
        float mx = fmaxf(s0[v], s1[v]);
#pragma unroll
        for (int off = 8; off >= 1; off >>= 1)
          mx = fmaxf(mx, __shfl_xor(mx, off, 16));
        const float mnew  = fmaxf(mrow[v], mx);
        const float alpha = __expf(mrow[v] - mnew);
        const float p0 = __expf(s0[v] - mnew);
        const float p1 = __expf(s1[v] - mnew);
        s0[v] = p0; s1[v] = p1;
        float rs = p0 + p1;
#pragma unroll
        for (int off = 8; off >= 1; off >>= 1)
          rs += __shfl_xor(rs, off, 16);
        lrow[v] = lrow[v] * alpha + rs;
        mrow[v] = mnew;
        acc0[v] *= alpha; acc1[v] *= alpha; acc2[v] *= alpha; acc3[v] *= alpha;
      }

      // C-fragment -> A-fragment relayout via wave-private LDS (DS ops from
      // one wave are in-order; no barrier needed).
#pragma unroll
      for (int v = 0; v < 8; ++v) {
        Ps[w][(v + 8 * lh) * 32 + n]      = (__bf16)s0[v];
        Ps[w][(v + 8 * lh) * 32 + 16 + n] = (__bf16)s1[v];
      }
      const v16bf aP = load_a_frag_u(Ps[w], 32, lane);

      const uint32_t* vu = (const uint32_t*)Vt[buf];
      acc0 = wmma_bf16(aP, load_b_nmaj_u(vu + 0 * 256, 16, lane), acc0);
      acc1 = wmma_bf16(aP, load_b_nmaj_u(vu + 1 * 256, 16, lane), acc1);
      acc2 = wmma_bf16(aP, load_b_nmaj_u(vu + 2 * 256, 16, lane), acc2);
      acc3 = wmma_bf16(aP, load_b_nmaj_u(vu + 3 * 256, 16, lane), acc3);
    }
    __syncthreads();
  }

  // Normalize and write context in [B,S,E] bf16 (col = h*D + d).
#pragma unroll
  for (int v = 0; v < 8; ++v) {
    const float inv = 1.0f / lrow[v];
    const int r = bb * Sn + q0 + v + 8 * lh;
    const size_t base = (size_t)r * En + hh * Dn;
    Cb[base + 0  + n] = (__bf16)(acc0[v] * inv);
    Cb[base + 16 + n] = (__bf16)(acc1[v] * inv);
    Cb[base + 32 + n] = (__bf16)(acc2[v] * inv);
    Cb[base + 48 + n] = (__bf16)(acc3[v] * inv);
  }
}

// ---------------------------------------------------------------------------
// Kernel 3: Out = ctx(bf16) @ Wo + bo, f32 output. A tile double-buffered
// async (1 async per wave per k-step -> wait asynccnt<=1 keeps next in flight).
// ---------------------------------------------------------------------------
__global__ __launch_bounds__(256) void out_proj_kernel(
    const __bf16* __restrict__ Xbf, const float* __restrict__ W,
    const float* __restrict__ bias, float* __restrict__ Out) {
  __shared__ __align__(16) __bf16 As[2][64 * 32];
  __shared__ __align__(16) __bf16 BsT[64 * 32];
  const int tid  = threadIdx.x;
  const int lane = tid & 31, w = tid >> 5;
  const int rbase = blockIdx.y * 64;
  const int cbase = blockIdx.x * 64;
  const int rsub  = (w & 3) << 4;
  const int nsub  = (w >> 2) << 5;
  const int rrA = tid >> 2, qA = tid & 3;

  v8f acc0 = {}, acc1 = {};
  const __bf16* Arow = Xbf + (size_t)(rbase + rrA) * En + qA * 8;

  copy16_g2l(Arow + 0, As[0] + rrA * 32 + qA * 8);

  for (int kIdx = 0; kIdx < En / 32; ++kIdx) {
    const int k0  = kIdx * 32;
    const int buf = kIdx & 1;
    const bool hasNext = (kIdx + 1) < (En / 32);
    if (hasNext)
      copy16_g2l(Arow + k0 + 32, As[buf ^ 1] + rrA * 32 + qA * 8);
    // B tile transposed (sync, overlaps the in-flight async A copy).
#pragma unroll
    for (int it = 0; it < 4; ++it) {
      const int i = tid + it * 256;
      const int nn = i & 63, kp = i >> 6;
      const float w0 = W[(size_t)(k0 + 2 * kp) * En + cbase + nn];
      const float w1 = W[(size_t)(k0 + 2 * kp + 1) * En + cbase + nn];
      ((uint32_t*)BsT)[nn * 16 + kp] = pk_bf16(w0, w1);
    }
    if (hasNext) wait_async_le1(); else wait_async_le0();
    __syncthreads();
    const v16bf a  = load_a_frag_u(As[buf] + rsub * 32, 32, lane);
    const v16bf b0 = load_b_nmaj_u((const uint32_t*)BsT + nsub * 16, 16, lane);
    const v16bf b1 = load_b_nmaj_u((const uint32_t*)BsT + (nsub + 16) * 16, 16, lane);
    acc0 = wmma_bf16(a, b0, acc0);
    acc1 = wmma_bf16(a, b1, acc1);
    __syncthreads();
  }

  const int n = lane & 15, lh = lane >> 4;
#pragma unroll
  for (int v = 0; v < 8; ++v) {
    const int r = rbase + rsub + v + 8 * lh;
#pragma unroll
    for (int t = 0; t < 2; ++t) {
      const int c = cbase + nsub + t * 16 + n;
      Out[(size_t)r * En + c] = (t ? acc1 : acc0)[v] + bias[c];
    }
  }
}

// ---------------------------------------------------------------------------
extern "C" void kernel_launch(void* const* d_in, const int* in_sizes, int n_in,
                              void* d_out, int out_size, void* d_ws, size_t ws_size,
                              hipStream_t stream) {
  (void)in_sizes; (void)n_in; (void)out_size; (void)ws_size;

  const float* query = (const float*)d_in[0];
  const float* key   = (const float*)d_in[1];
  const float* value = (const float*)d_in[2];
  const float* Wq    = (const float*)d_in[3];
  const float* Wk    = (const float*)d_in[4];
  const float* Wv    = (const float*)d_in[5];
  const float* Wo    = (const float*)d_in[6];
  const float* bq    = (const float*)d_in[7];
  const float* bk    = (const float*)d_in[8];
  const float* bv    = (const float*)d_in[9];
  const float* bo    = (const float*)d_in[10];
  // d_in[11] = attn_mask: causal, implemented analytically in-kernel.
  float* out = (float*)d_out;

  char* ws = (char*)d_ws;
  const size_t seg = (size_t)Bn * Sn * En * sizeof(__bf16);  // 8 MiB
  __bf16* Qb  = (__bf16*)(ws);
  __bf16* Kb  = (__bf16*)(ws + seg);
  __bf16* VbT = (__bf16*)(ws + 2 * seg);  // [B,H,D,S]
  __bf16* Cb  = (__bf16*)(ws + 3 * seg);

  const dim3 gproj(En / 64, Rn / 64, 1);  // 16 x 64 blocks
  // Both scalings (D^-0.5 at projection, 1/sqrt(D) in attention) fold to 1/D.
  qkv_proj_kernel<0><<<gproj, 256, 0, stream>>>(query, Wq, bq, Qb, 1.0f / 64.0f);
  qkv_proj_kernel<0><<<gproj, 256, 0, stream>>>(key,   Wk, bk, Kb, 1.0f);
  qkv_proj_kernel<1><<<gproj, 256, 0, stream>>>(value, Wv, bv, VbT, 1.0f);

  attn_kernel<<<dim3(Sn / 128, Bn * Hn), 256, 0, stream>>>(Qb, Kb, VbT, Cb);

  out_proj_kernel<<<gproj, 256, 0, stream>>>(Cb, Wo, bo, out);
}
